// GAT_79053168050934
// MI455X (gfx1250) — matrix operand order repaired and checked by reading, not compiled, not run on hardware
//
#include <hip/hip_runtime.h>
#include <math.h>

#define NNODES 50000
#define NHEADS 4
#define NEG_SLOPE 0.2f

typedef __attribute__((ext_vector_type(16))) _Float16 v16h;
typedef __attribute__((ext_vector_type(8)))  _Float16 v8h;
typedef __attribute__((ext_vector_type(8)))  float    v8f;

// ---------- monotone float<->uint encoding for integer atomicMax ----------
__device__ __forceinline__ unsigned enc_f32(float f) {
    unsigned u = __float_as_uint(f);
    return (u & 0x80000000u) ? ~u : (u ^ 0x80000000u);
}
__device__ __forceinline__ float dec_f32(unsigned e) {
    unsigned u = (e & 0x80000000u) ? (e ^ 0x80000000u) : ~e;
    return __uint_as_float(u);
}

// ---------- fp32 -> fp16 conversion ----------
__global__ void cvt_f32_to_f16(const float* __restrict__ in, _Float16* __restrict__ out, int n) {
    int i = blockIdx.x * blockDim.x + threadIdx.x;
    if (i < n) out[i] = (_Float16)in[i];
}

// W [K, Nout] fp32 -> Wt [Nout, K] fp16 (transposed so B tiles load contiguously)
__global__ void cvt_w_transpose(const float* __restrict__ W, _Float16* __restrict__ Wt,
                                int K, int Nout) {
    int i = blockIdx.x * blockDim.x + threadIdx.x;
    if (i < K * Nout) {
        int nn = i / K, k = i % K;
        Wt[i] = (_Float16)W[k * Nout + nn];
    }
}

// ---------- WMMA GEMM: C[M,Nout] = A[M,K] * Bt[Nout,K]^T ----------
// One wave per 16x16 output tile, K stepped by 32 with v_wmma_f32_16x16x32_f16.
__global__ void wmma_gemm_f16(const _Float16* __restrict__ A, const _Float16* __restrict__ Bt,
                              float* __restrict__ Cm, int Nout, int K, int numTiles) {
    int wv = (blockIdx.x * blockDim.x + threadIdx.x) >> 5;
    if (wv >= numTiles) return;                       // wave-uniform: EXEC stays all-ones
    int lane = threadIdx.x & 31;
    int r    = lane & 15;
    int side = lane >> 4;                             // which half of the wave
    int tN   = Nout >> 4;
    int mT   = wv / tN, nT = wv % tN;

    // A lane layout: element i -> K = (i<8?0:16) + side*8 + (i&7)
    const _Float16* arow = A + (size_t)(mT * 16 + r) * K + side * 8;
    // B lane layout: element i -> K = (i<8?0:16) + side*8 + (i&7)
    const _Float16* brow = Bt + (size_t)(nT * 16 + r) * K + side * 8;

    v8f acc = {};
    for (int k0 = 0; k0 < K; k0 += 32) {
        v8h alo = *(const v8h*)(arow + k0);           // K: side*8 + 0..7
        v8h ahi = *(const v8h*)(arow + k0 + 16);      // K: side*8 + 16..23
        v8h blo = *(const v8h*)(brow + k0);           // K: side*8 + 0..7
        v8h bhi = *(const v8h*)(brow + k0 + 16);      // K: side*8 + 16..23
        v16h a, b;
#pragma unroll
        for (int i = 0; i < 8; ++i) {
            a[i] = alo[i]; a[i + 8] = ahi[i];
            b[i] = blo[i]; b[i + 8] = bhi[i];
        }
        acc = __builtin_amdgcn_wmma_f32_16x16x32_f16(false, a, false, b,
                                                     (short)0, acc, false, false);
    }
    // D layout: VGPR v -> row (v + side*8), col r
    float* crow = Cm + (size_t)(mT * 16 + side * 8) * Nout + nT * 16 + r;
#pragma unroll
    for (int v = 0; v < 8; ++v) crow[(size_t)v * Nout] = acc[v];
}

// ---------- attention logits: alpha_s/d[n,h] = sum_c xp[n,h,c] * a_{s,d}[h,c] ----------
__global__ void attn_coeffs(const float* __restrict__ xp, const float* __restrict__ a_s,
                            const float* __restrict__ a_d, int C,
                            float* __restrict__ als, float* __restrict__ ald) {
    int i = blockIdx.x * blockDim.x + threadIdx.x;
    if (i >= NNODES * NHEADS) return;
    int n = i / NHEADS, h = i % NHEADS;
    const float* row = xp + (size_t)n * NHEADS * C + (size_t)h * C;
    const float* vs = a_s + h * C;
    const float* vd = a_d + h * C;
    float ss = 0.f, sd = 0.f;
    for (int c = 0; c < C; ++c) { float v = row[c]; ss += v * vs[c]; sd += v * vd[c]; }
    als[i] = ss; ald[i] = sd;
}

__device__ __forceinline__ void edge_src_dst(const long long* ei, int E, int e,
                                             int& src, int& dst) {
    if (e < E) { src = (int)ei[e]; dst = (int)ei[E + e]; }
    else       { src = dst = e - E; }                 // appended self-loops
}

__device__ __forceinline__ float leaky(float v) { return v > 0.f ? v : NEG_SLOPE * v; }

// ---------- pass 1: segment max ----------
__global__ void edge_max(const long long* __restrict__ ei, int E, int Etot,
                         const float* __restrict__ als, const float* __restrict__ ald,
                         unsigned* __restrict__ mEnc) {
    int i = blockIdx.x * blockDim.x + threadIdx.x;
    if (i >= Etot * NHEADS) return;
    int e = i / NHEADS, h = i % NHEADS;
    int src, dst; edge_src_dst(ei, E, e, src, dst);
    float v = leaky(als[src * NHEADS + h] + ald[dst * NHEADS + h]);
    atomicMax(&mEnc[dst * NHEADS + h], enc_f32(v));
}

// ---------- pass 2: segment sum of exp(e - m) ----------
__global__ void edge_sum(const long long* __restrict__ ei, int E, int Etot,
                         const float* __restrict__ als, const float* __restrict__ ald,
                         const unsigned* __restrict__ mEnc, float* __restrict__ denom) {
    int i = blockIdx.x * blockDim.x + threadIdx.x;
    if (i >= Etot * NHEADS) return;
    int e = i / NHEADS, h = i % NHEADS;
    int src, dst; edge_src_dst(ei, E, e, src, dst);
    float v = leaky(als[src * NHEADS + h] + ald[dst * NHEADS + h]);
    float m = dec_f32(mEnc[dst * NHEADS + h]);
    atomicAdd(&denom[dst * NHEADS + h], __expf(v - m));
}

// ---------- pass 3: weighted scatter, one wave per edge ----------
__global__ void edge_scatter(const long long* __restrict__ ei, int E, int Etot,
                             const float* __restrict__ als, const float* __restrict__ ald,
                             const unsigned* __restrict__ mEnc, const float* __restrict__ denom,
                             const float* __restrict__ xp, float* __restrict__ agg, int C) {
    int wv = (blockIdx.x * blockDim.x + threadIdx.x) >> 5;
    if (wv >= Etot) return;
    int lane = threadIdx.x & 31;
    int src, dst; edge_src_dst(ei, E, wv, src, dst);
    int HC = NHEADS * C;
    const float* xs = xp + (size_t)src * HC;
    float* od = agg + (size_t)dst * HC;
    __builtin_prefetch(xs + lane, 0, 1);              // global_prefetch_b8
    for (int idx = lane; idx < HC; idx += 32) {
        int h = idx / C;
        float v = leaky(als[src * NHEADS + h] + ald[dst * NHEADS + h]);
        float m = dec_f32(mEnc[dst * NHEADS + h]);
        float w = __expf(v - m) / denom[dst * NHEADS + h];
        atomicAdd(&od[idx], w * xs[idx]);
    }
}

// ---------- head mean + bias + relu (layers 1,2) ----------
__global__ void head_mean_relu(const float* __restrict__ agg, const float* __restrict__ bias,
                               float* __restrict__ out, int C) {
    int i = blockIdx.x * blockDim.x + threadIdx.x;
    if (i >= NNODES * C) return;
    int n = i / C, c = i % C;
    const float* row = agg + (size_t)n * NHEADS * C + c;
    float s = 0.f;
#pragma unroll
    for (int h = 0; h < NHEADS; ++h) s += row[(size_t)h * C];
    out[i] = fmaxf(s * (1.0f / NHEADS) + bias[c], 0.f);
}

// ---------- layer 3: head mean + bias + relu + log_softmax, one wave per node ----------
__global__ void final_logsoftmax(const float* __restrict__ agg, const float* __restrict__ bias,
                                 float* __restrict__ out, int C) {
    int wv = (blockIdx.x * blockDim.x + threadIdx.x) >> 5;
    if (wv >= NNODES) return;
    int lane = threadIdx.x & 31;
    const float* row = agg + (size_t)wv * NHEADS * C;
    int c0 = lane, c1 = lane + 32;
    float v0 = -INFINITY, v1 = -INFINITY;
    if (c0 < C) {
        float s = 0.f;
#pragma unroll
        for (int h = 0; h < NHEADS; ++h) s += row[(size_t)h * C + c0];
        v0 = fmaxf(s * (1.0f / NHEADS) + bias[c0], 0.f);
    }
    if (c1 < C) {
        float s = 0.f;
#pragma unroll
        for (int h = 0; h < NHEADS; ++h) s += row[(size_t)h * C + c1];
        v1 = fmaxf(s * (1.0f / NHEADS) + bias[c1], 0.f);
    }
    float mx = fmaxf(v0, v1);
    for (int off = 16; off; off >>= 1) mx = fmaxf(mx, __shfl_xor(mx, off, 32));
    float se = (c0 < C ? __expf(v0 - mx) : 0.f) + (c1 < C ? __expf(v1 - mx) : 0.f);
    for (int off = 16; off; off >>= 1) se += __shfl_xor(se, off, 32);
    float lse = __logf(se) + mx;
    if (c0 < C) out[(size_t)wv * C + c0] = v0 - lse;
    if (c1 < C) out[(size_t)wv * C + c1] = v1 - lse;
}

extern "C" void kernel_launch(void* const* d_in, const int* in_sizes, int n_in,
                              void* d_out, int out_size, void* d_ws, size_t ws_size,
                              hipStream_t stream) {
    const float*      x  = (const float*)d_in[0];
    const long long*  ei = (const long long*)d_in[1];
    const float* W [3] = {(const float*)d_in[2], (const float*)d_in[6],  (const float*)d_in[10]};
    const float* As[3] = {(const float*)d_in[3], (const float*)d_in[7],  (const float*)d_in[11]};
    const float* Ad[3] = {(const float*)d_in[4], (const float*)d_in[8],  (const float*)d_in[12]};
    const float* Bb[3] = {(const float*)d_in[5], (const float*)d_in[9],  (const float*)d_in[13]};
    const int Fin [3] = {128, 64, 64};
    const int Cdim[3] = {64, 64, 40};

    const int E    = in_sizes[1] / 2;       // 800000
    const int Etot = E + NNODES;            // + self loops

    // ---- workspace carve (256B aligned) ----
    char* wsp = (char*)d_ws;
    size_t off = 0;
    auto carve = [&](size_t bytes) -> void* {
        void* p = wsp + off;
        off += (bytes + 255) & ~(size_t)255;
        return p;
    };
    _Float16* x16  = (_Float16*)carve((size_t)NNODES * 128 * sizeof(_Float16));
    _Float16* Wt16 = (_Float16*)carve((size_t)256 * 128 * sizeof(_Float16));
    float*    xp   = (float*)   carve((size_t)NNODES * 256 * sizeof(float));
    float*    agg  = (float*)   carve((size_t)NNODES * 256 * sizeof(float));
    float*    als  = (float*)   carve((size_t)NNODES * NHEADS * sizeof(float));
    float*    ald  = (float*)   carve((size_t)NNODES * NHEADS * sizeof(float));
    unsigned* mEnc = (unsigned*)carve((size_t)NNODES * NHEADS * sizeof(unsigned));
    float*    den  = (float*)   carve((size_t)NNODES * NHEADS * sizeof(float));
    float*    hbuf = (float*)   carve((size_t)NNODES * 64 * sizeof(float));
    (void)ws_size; (void)n_in; (void)out_size;

    auto cdiv = [](int a, int b) { return (a + b - 1) / b; };
    const int TB = 256;

    const float* cur = x;
    for (int l = 0; l < 3; ++l) {
        const int K = Fin[l], C = Cdim[l], Hout = NHEADS * C;

        cvt_f32_to_f16<<<cdiv(NNODES * K, TB), TB, 0, stream>>>(cur, x16, NNODES * K);
        cvt_w_transpose<<<cdiv(K * Hout, TB), TB, 0, stream>>>(W[l], Wt16, K, Hout);

        const int numTiles = (NNODES / 16) * (Hout / 16);
        wmma_gemm_f16<<<cdiv(numTiles, TB / 32), TB, 0, stream>>>(x16, Wt16, xp, Hout, K, numTiles);

        attn_coeffs<<<cdiv(NNODES * NHEADS, TB), TB, 0, stream>>>(xp, As[l], Ad[l], C, als, ald);

        hipMemsetAsync(mEnc, 0, (size_t)NNODES * NHEADS * sizeof(unsigned), stream);
        hipMemsetAsync(den,  0, (size_t)NNODES * NHEADS * sizeof(float),    stream);
        hipMemsetAsync(agg,  0, (size_t)NNODES * Hout  * sizeof(float),     stream);

        edge_max<<<cdiv(Etot * NHEADS, TB), TB, 0, stream>>>(ei, E, Etot, als, ald, mEnc);
        edge_sum<<<cdiv(Etot * NHEADS, TB), TB, 0, stream>>>(ei, E, Etot, als, ald, mEnc, den);
        edge_scatter<<<cdiv(Etot, TB / 32), TB, 0, stream>>>(ei, E, Etot, als, ald, mEnc, den,
                                                             xp, agg, C);

        if (l < 2) {
            head_mean_relu<<<cdiv(NNODES * C, TB), TB, 0, stream>>>(agg, Bb[l], hbuf, C);
            cur = hbuf;
        } else {
            final_logsoftmax<<<cdiv(NNODES, TB / 32), TB, 0, stream>>>(agg, Bb[l],
                                                                       (float*)d_out, C);
        }
    }
}
